// PerspectiveNet768x2_69372311765153
// MI455X (gfx1250) — matrix operand order repaired and checked by reading, not compiled, not run on hardware
//
#include <hip/hip_runtime.h>
#include <hip/hip_bf16.h>

// PerspectiveNet768x2 (NNUE-style) fused kernel for MI455X / gfx1250.
//
// Roofline reasoning:
//  - Feature transformer is a sparse gather-sum (B*64 rows of 4KB = ~4.3GB of
//    gather reads from 31.5MB of tables -> L2-resident, L2-BW bound). Dense
//    WMMA formulation would be ~129 TFLOP of wasted work; we use coalesced
//    b128 gathers + register accumulation instead.
//  - Output layer ([B,2048] x [2048] dot) maps onto v_wmma_f32_16x16x32_f16:
//    16 positions per workgroup become the 16 A-rows, W_out is replicated
//    across the 16 B-columns so every C column holds the 16 results.

typedef __attribute__((ext_vector_type(16))) _Float16 v16h;
typedef __attribute__((ext_vector_type(8)))  _Float16 v8h;
typedef __attribute__((ext_vector_type(4)))  _Float16 v4h;
typedef __attribute__((ext_vector_type(8)))  float    v8f;

#define HID        1024
#define NACT       32
#define POS_PER_WG 16
#define TPB        256
#define ROW        (2 * HID + 8)   // padded f16 row stride (halves): bank-spread

__device__ inline v4h clip_sq_4(float4 v) {
    v4h o;
    float x;
    x = fminf(fmaxf(v.x, 0.f), 1.f); o[0] = (_Float16)(x * x);
    x = fminf(fmaxf(v.y, 0.f), 1.f); o[1] = (_Float16)(x * x);
    x = fminf(fmaxf(v.z, 0.f), 1.f); o[2] = (_Float16)(x * x);
    x = fminf(fmaxf(v.w, 0.f), 1.f); o[3] = (_Float16)(x * x);
    return o;
}

__global__ __launch_bounds__(TPB)
void nnue_fused_kernel(const int* __restrict__ fw,
                       const int* __restrict__ fb,
                       const unsigned char* __restrict__ stm,   // jnp bool: 1 byte
                       const float* __restrict__ Ww,
                       const float* __restrict__ bw,
                       const float* __restrict__ Wb,
                       const float* __restrict__ bb,
                       const float* __restrict__ Wo,
                       const float* __restrict__ bo,
                       float* __restrict__ out,
                       int Bsz)
{
    __shared__ _Float16 act[POS_PER_WG * ROW];   // stm-ordered activations, f16
    __shared__ _Float16 wout[2 * HID];           // W_out as f16

    const int tid  = threadIdx.x;
    const int base = blockIdx.x * POS_PER_WG;

    // Stage W_out into LDS as f16 (8 KB of reads, once per workgroup).
    #pragma unroll
    for (int e = tid; e < 2 * HID; e += TPB)
        wout[e] = (_Float16)Wo[e];

    // Thread t owns hidden elements [4t, 4t+3] of each perspective.
    const float4 bw4 = *(const float4*)(bw + tid * 4);
    const float4 bb4 = *(const float4*)(bb + tid * 4);

    // ---- Phase 1: feature transformer (gather-sum), one position at a time ----
    for (int p = 0; p < POS_PER_WG; ++p) {
        const int pos = min(base + p, Bsz - 1);
        float4 hw = bw4;
        float4 hb = bb4;
        const int* iw = fw + (size_t)pos * NACT;
        const int* ib = fb + (size_t)pos * NACT;

        #pragma unroll 4
        for (int n = 0; n < NACT; ++n) {
            const int i0 = iw[n];                 // uniform -> scalar load/branch
            if (i0 >= 0) {
                const float4 r = *(const float4*)(Ww + (size_t)i0 * HID + tid * 4);
                hw.x += r.x; hw.y += r.y; hw.z += r.z; hw.w += r.w;
            }
            const int i1 = ib[n];
            if (i1 >= 0) {
                const float4 r = *(const float4*)(Wb + (size_t)i1 * HID + tid * 4);
                hb.x += r.x; hb.y += r.y; hb.z += r.z; hb.w += r.w;
            }
        }

        // stm-ordered concat + clipped-square, store f16 to LDS.
        const bool white_stm = stm[pos] != 0;
        const int offW = white_stm ? 0 : HID;
        const int offB = white_stm ? HID : 0;
        *(v4h*)(act + p * ROW + offW + tid * 4) = clip_sq_4(hw);
        *(v4h*)(act + p * ROW + offB + tid * 4) = clip_sq_4(hb);
    }

    __syncthreads();

    // ---- Phase 2: output layer via WMMA (wave 0 only; EXEC all-ones there) ----
    if (tid < 32) {
        const int lane = tid;
        const int m    = lane & 15;                   // A row = position in tile
        const int sA   = (lane >= 16) ? 8  : 0;       // 16-bit A 16x32 lane layout
        const int sB   = (lane >= 16) ? 8  : 0;       // 16-bit B 32x16 lane layout
        const _Float16* arow = act + m * ROW;

        v8f c = {};
        #pragma unroll 4
        for (int kb = 0; kb < 2 * HID; kb += 32) {
            // A fragment: elems 0..7 -> K = kb+sA+0..7 ; elems 8..15 -> K = kb+16+sA+0..7
            const v8h alo = *(const v8h*)(arow + kb + sA);
            const v8h ahi = *(const v8h*)(arow + kb + sA + 16);
            // B fragment: elems 0..7 -> K = kb+sB+0..7 ; elems 8..15 -> K = kb+16+sB+0..7 (replicated across all 16 cols)
            const v8h blo = *(const v8h*)(wout + kb + sB);
            const v8h bhi = *(const v8h*)(wout + kb + sB + 16);
            const v16h a = __builtin_shufflevector(alo, ahi,
                0, 1, 2, 3, 4, 5, 6, 7, 8, 9, 10, 11, 12, 13, 14, 15);
            const v16h b = __builtin_shufflevector(blo, bhi,
                0, 1, 2, 3, 4, 5, 6, 7, 8, 9, 10, 11, 12, 13, 14, 15);
            c = __builtin_amdgcn_wmma_f32_16x16x32_f16(
                    /*neg_a=*/false, a, /*neg_b=*/false, b,
                    /*c_mod=*/(short)0, c, /*reuse_a=*/false, /*reuse_b=*/false);
        }

        // C layout: VGPR r holds M=r (lanes 0-15) / M=8+r (lanes 16-31); every
        // column is identical since B was replicated. Lane 0 / lane 16 emit.
        const float bias = bo[0];
        if (lane == 0) {
            #pragma unroll
            for (int r = 0; r < 8; ++r) {
                const int pos = base + r;
                if (pos < Bsz) out[pos] = c[r] + bias;
            }
        } else if (lane == 16) {
            #pragma unroll
            for (int r = 0; r < 8; ++r) {
                const int pos = base + 8 + r;
                if (pos < Bsz) out[pos] = c[r] + bias;
            }
        }
    }
}

extern "C" void kernel_launch(void* const* d_in, const int* in_sizes, int n_in,
                              void* d_out, int out_size, void* d_ws, size_t ws_size,
                              hipStream_t stream) {
    const int*           fw  = (const int*)d_in[0];            // [B, 32] int32
    const int*           fb  = (const int*)d_in[1];            // [B, 32] int32
    const unsigned char* stm = (const unsigned char*)d_in[2];  // [B, 1] bool (1 byte)
    const float*         Ww  = (const float*)d_in[3];          // [3840, 1024]
    const float*         bw  = (const float*)d_in[4];          // [1024]
    const float*         Wb  = (const float*)d_in[5];          // [3840, 1024]
    const float*         bb  = (const float*)d_in[6];          // [1024]
    const float*         Wo  = (const float*)d_in[7];          // [1, 2048]
    const float*         bo  = (const float*)d_in[8];          // [1]
    float*               out = (float*)d_out;                  // [B, 1]

    const int Bsz    = in_sizes[0] / NACT;
    const int blocks = (Bsz + POS_PER_WG - 1) / POS_PER_WG;

    hipLaunchKernelGGL(nnue_fused_kernel, dim3(blocks), dim3(TPB), 0, stream,
                       fw, fb, stm, Ww, bw, Wb, bb, Wo, bo, out, Bsz);
}
